// AlternatingForecastModel_82617990906594
// MI455X (gfx1250) — compile-verified, matrix operand order checked by reading, not run on hardware
//
#include <hip/hip_runtime.h>

// ---------------- problem constants ----------------
#define BB 64
#define SS 1344
#define FF 16
#define HH 256
#define WEEK 672
#define G4H 1024            // 4*H
#define KT 8                // K tiles of 32 over 256
#define TILE_ELEMS 512      // 32x16 bf16 elements per WMMA B tile
#define TPB 512             // 16 waves per workgroup -> 4 waves/SIMD -> 256 VGPRs/wave
#define NTW 4               // N-tiles per wave (64 tiles / 16 waves)

// h buffers: rows padded to 264 bf16 (528B = 16B multiple, bank-friendly)
#define HSTRIDE 264
#define HSTRIDE_DW (HSTRIDE / 2)

// tile table in workspace: [Wih0: 64][Whh0: 64*8][Wih1: 64*8][Whh1: 64*8]
#define T_IH0 0
#define T_HH0 64
#define T_IH1 (64 + 512)
#define T_HH1 (64 + 1024)
#define N_TILES_TOTAL (64 + 1536)
#define W_ELEMS (N_TILES_TOTAL * TILE_ELEMS)   // 819200 bf16 elements

// f32 aux region (after weights): b0[1024] b1[1024] wout[256] bout[1]
#define FB_B0   0
#define FB_B1   1024
#define FB_WOUT 2048
#define FB_BOUT 2304

typedef __attribute__((ext_vector_type(16))) __bf16 v16bf;
typedef __attribute__((ext_vector_type(8)))  float  v8f;

// global-address-space pointers so weight streams lower to global_load_b128
// (LOADcnt only), not flat_load (LOADcnt+DScnt coupling with LDS traffic)
typedef const __attribute__((address_space(1))) unsigned short* gwptr;
typedef const __attribute__((address_space(1))) v16bf*          gvptr;

union FragBF { v16bf v; unsigned int u[8]; };

__device__ __forceinline__ unsigned short f2bf(float x) {
  unsigned int u = __float_as_uint(x);
  unsigned int r = (u + 0x7FFFu + ((u >> 16) & 1u)) >> 16;   // RNE
  return (unsigned short)r;
}
__device__ __forceinline__ float bf2f(unsigned short s) {
  return __uint_as_float(((unsigned int)s) << 16);
}
__device__ __forceinline__ float sigm(float x) {
  return 1.0f / (1.0f + __expf(-x));
}

__device__ __forceinline__ v8f wmma_bf16(v16bf a, v16bf b, v8f c) {
  return __builtin_amdgcn_wmma_f32_16x16x32_bf16(false, a, false, b, (short)0, c,
                                                 false, false);
}

// B fragment: pre-swizzled, contiguous 32B per lane, global addrspace
__device__ __forceinline__ v16bf load_btile(gwptr w, int tile, int lane) {
  return *(gvptr)(w + (size_t)tile * TILE_ELEMS + (size_t)lane * 16);
}

// A fragment from LDS (row-major bf16, strideDW dwords per row), per ISA
// 16-bit A 16x32 layout: lane%16 = M, lane/16 selects K half, regs hold K pairs.
__device__ __forceinline__ v16bf load_afrag(const unsigned short* hl, int strideDW,
                                            int mrow, int g, int kt) {
  FragBF f;
  const unsigned int* base = (const unsigned int*)hl;
  int rowd = mrow * strideDW + kt * 16 + 4 * g;  // dword offset
  #pragma unroll
  for (int r = 0; r < 8; ++r) {
    int kd = (r < 4) ? r : (4 + r);              // koff/2 = r + (r>=4?4:0)
    f.u[r] = base[rowd + kd];
  }
  return f.v;
}

// ---------------- prep kernel: repack weights to bf16 WMMA tiles ----------------
__global__ void lstm_prep_kernel(const float* __restrict__ Wih0, const float* __restrict__ Whh0,
                                 const float* __restrict__ Wih1, const float* __restrict__ Whh1,
                                 const float* __restrict__ bih0, const float* __restrict__ bhh0,
                                 const float* __restrict__ bih1, const float* __restrict__ bhh1,
                                 const float* __restrict__ Wout, const float* __restrict__ bout,
                                 unsigned short* __restrict__ wbf, float* __restrict__ fbuf) {
  int idx = blockIdx.x * blockDim.x + threadIdx.x;
  if (idx < W_ELEMS) {
    int tile = idx / TILE_ELEMS;
    int e    = idx % TILE_ELEMS;
    int lane = e >> 4, pos = e & 15;
    int r = pos >> 1, hb = pos & 1;
    int g = lane >> 4, n_in = lane & 15;
    int koff = 2 * r + ((r >= 4) ? 8 : 0) + 8 * g + hb;
    const float* W; int nt, kt, ksrc;
    if (tile < T_HH0)      { W = Wih0; nt = tile;                 kt = 0;        ksrc = FF + 1; }
    else if (tile < T_IH1) { int t2 = tile - T_HH0; W = Whh0; nt = t2 >> 3; kt = t2 & 7; ksrc = HH; }
    else if (tile < T_HH1) { int t2 = tile - T_IH1; W = Wih1; nt = t2 >> 3; kt = t2 & 7; ksrc = HH; }
    else                   { int t2 = tile - T_HH1; W = Whh1; nt = t2 >> 3; kt = t2 & 7; ksrc = HH; }
    int K = kt * 32 + koff;
    int N = nt * 16 + n_in;
    float v = (K < ksrc) ? W[(size_t)N * ksrc + K] : 0.0f;
    wbf[idx] = f2bf(v);
  } else {
    int j = idx - W_ELEMS;
    if (j < G4H)                 fbuf[FB_B0 + j] = bih0[j] + bhh0[j];
    else if (j < 2 * G4H)        fbuf[FB_B1 + (j - G4H)] = bih1[j - G4H] + bhh1[j - G4H];
    else if (j < 2 * G4H + HH)   fbuf[FB_WOUT + (j - 2 * G4H)] = Wout[j - 2 * G4H];
    else if (j == 2 * G4H + HH)  fbuf[FB_BOUT] = bout[0];
  }
}

// ---------------- persistent LSTM kernel: 1 workgroup = 16 batch rows ----------------
__global__ void __launch_bounds__(TPB, 1)
lstm_scan_kernel(const float* __restrict__ x, const unsigned short* __restrict__ wbf,
                 const float* __restrict__ fbuf, float* __restrict__ out) {
  extern __shared__ char smem[];
  float*          gates = (float*)smem;                                 // 16*1024 f32
  unsigned short* h1l   = (unsigned short*)(smem + 16 * 1024 * 4);      // 16*HSTRIDE bf16
  unsigned short* h2l   = h1l + 16 * HSTRIDE;                           // 16*HSTRIDE bf16
  unsigned short* inl   = h2l + 16 * HSTRIDE;                           // 16*34 bf16 (padded)
  float*          pout  = (float*)(inl + 16 * 34);                      // 16 f32

  const int tid  = threadIdx.x;
  const int lane = tid & 31;
  const int wave = tid >> 5;                  // 0..15
  const int b0   = blockIdx.x * 16;

  const int g    = lane >> 4;                 // K-half select for WMMA fragments
  const int mrow = lane & 15;                 // M row (A) / N col (C,D)
  const int nt0  = wave * NTW;                // this wave's first N-tile (of 64)

  // pointwise ownership: fixed (m, j..j+7) per thread across all steps
  const int pm = tid >> 5;                    // 0..15
  const int pj = (tid & 31) * 8;              // 0..248 step 8
  float c1[8], c2[8];
  #pragma unroll
  for (int u = 0; u < 8; ++u) { c1[u] = 0.f; c2[u] = 0.f; }

  // hoisted per-lane constants
  float bias0[NTW], bias1[NTW];
  #pragma unroll
  for (int q = 0; q < NTW; ++q) {
    bias0[q] = fbuf[FB_B0 + (nt0 + q) * 16 + mrow];
    bias1[q] = fbuf[FB_B1 + (nt0 + q) * 16 + mrow];
  }
  const float boutv = fbuf[FB_BOUT];
  float wout_r[8];
  #pragma unroll
  for (int k = 0; k < 8; ++k) wout_r[k] = fbuf[FB_WOUT + lane * 8 + k];

  // global-addrspace view of the weight tile table
  gwptr wgl = (gwptr)wbf;

  // init LDS state
  for (int i = tid; i < 16 * HSTRIDE; i += TPB) { h1l[i] = 0; h2l[i] = 0; }
  for (int i = tid; i < 16 * 34;      i += TPB) inl[i] = 0;
  if (tid < 16) pout[tid] = 0.0f;
  __syncthreads();

  for (int t = 0; t < SS; ++t) {
    // Optimization barrier: make the weight base pointer opaque per time step so
    // LICM cannot hoist the (loop-invariant) weight loads out of the t-loop and
    // spill them to scratch. Forces fresh L2-resident global_load_b128 per step.
    gwptr wblk = wgl;
    asm volatile("" : "+s"(wblk));

    // forcing mask: mask[i:i+WEEK]=0 for i in range(0,S,2*WEEK) if i+WEEK<=S; t==0 forced
    int phase = t % (2 * WEEK);
    bool use_orig = ((phase < WEEK) && ((t - phase) + WEEK <= SS)) || (t == 0);

    // ---- P0: build layer-0 input tile [wl, x1..x15, flag, 0-pad] ----
    if (tid < 256) {
      int m = tid >> 4, c = tid & 15;
      const float* xp = x + (size_t)(b0 + m) * SS * FF + (size_t)t * FF;
      float v = xp[c];
      if (t + 1 < SS) __builtin_prefetch(xp + FF, 0, 1);   // global_prefetch next step
      if (c == 0) {
        inl[m * 34 + 0]  = f2bf(use_orig ? v : pout[m]);
        inl[m * 34 + 16] = f2bf(use_orig ? 0.0f : 1.0f);
      } else {
        inl[m * 34 + c] = f2bf(v);
      }
    }
    __syncthreads();

    // ---- layer 0 GEMM: gates0 = bias0 + inp @ Wih0^T + h1_prev @ Whh0^T ----
    {
      v8f acc[NTW];
      v16bf fa = load_afrag(inl, 17, mrow, g, 0);
      #pragma unroll
      for (int q = 0; q < NTW; ++q) {
        v8f zc{};   // zero C -> first WMMA starts the chain (inline-0 SRC2)
        v16bf fb = load_btile(wblk, T_IH0 + nt0 + q, lane);
        acc[q] = wmma_bf16(fa, fb, zc);
      }
      #pragma unroll
      for (int kt = 0; kt < KT; ++kt) {
        v16bf ha = load_afrag(h1l, HSTRIDE_DW, mrow, g, kt);
        #pragma unroll
        for (int q = 0; q < NTW; ++q) {
          v16bf wb = load_btile(wblk, T_HH0 + (nt0 + q) * 8 + kt, lane);
          acc[q] = wmma_bf16(ha, wb, acc[q]);
        }
      }
      #pragma unroll
      for (int q = 0; q < NTW; ++q)
        #pragma unroll
        for (int v = 0; v < 8; ++v)
          gates[(v + 8 * g) * 1024 + (nt0 + q) * 16 + mrow] = acc[q][v] + bias0[q];
    }
    __syncthreads();

    // ---- pointwise layer 0 -> h1, c1 ----
    {
      float iv[8], fv[8], gv[8], ov[8], hv[8];
      *(float4*)&iv[0] = *(const float4*)&gates[pm * 1024 + pj];
      *(float4*)&iv[4] = *(const float4*)&gates[pm * 1024 + pj + 4];
      *(float4*)&fv[0] = *(const float4*)&gates[pm * 1024 + 256 + pj];
      *(float4*)&fv[4] = *(const float4*)&gates[pm * 1024 + 256 + pj + 4];
      *(float4*)&gv[0] = *(const float4*)&gates[pm * 1024 + 512 + pj];
      *(float4*)&gv[4] = *(const float4*)&gates[pm * 1024 + 512 + pj + 4];
      *(float4*)&ov[0] = *(const float4*)&gates[pm * 1024 + 768 + pj];
      *(float4*)&ov[4] = *(const float4*)&gates[pm * 1024 + 768 + pj + 4];
      #pragma unroll
      for (int u = 0; u < 8; ++u) {
        float c = sigm(fv[u]) * c1[u] + sigm(iv[u]) * tanhf(gv[u]);
        c1[u] = c;
        hv[u] = sigm(ov[u]) * tanhf(c);
      }
      uint4 packed;
      packed.x = (unsigned)f2bf(hv[0]) | ((unsigned)f2bf(hv[1]) << 16);
      packed.y = (unsigned)f2bf(hv[2]) | ((unsigned)f2bf(hv[3]) << 16);
      packed.z = (unsigned)f2bf(hv[4]) | ((unsigned)f2bf(hv[5]) << 16);
      packed.w = (unsigned)f2bf(hv[6]) | ((unsigned)f2bf(hv[7]) << 16);
      *(uint4*)&h1l[pm * HSTRIDE + pj] = packed;
    }
    __syncthreads();

    // ---- layer 1 GEMM: gates1 = bias1 + h1 @ Wih1^T + h2_prev @ Whh1^T ----
    {
      v8f acc[NTW];
      {
        v16bf a1 = load_afrag(h1l, HSTRIDE_DW, mrow, g, 0);
        #pragma unroll
        for (int q = 0; q < NTW; ++q) {
          v8f zc{};
          v16bf wb = load_btile(wblk, T_IH1 + (nt0 + q) * 8 + 0, lane);
          acc[q] = wmma_bf16(a1, wb, zc);
        }
        v16bf a2 = load_afrag(h2l, HSTRIDE_DW, mrow, g, 0);
        #pragma unroll
        for (int q = 0; q < NTW; ++q) {
          v16bf wb = load_btile(wblk, T_HH1 + (nt0 + q) * 8 + 0, lane);
          acc[q] = wmma_bf16(a2, wb, acc[q]);
        }
      }
      #pragma unroll
      for (int kt = 1; kt < KT; ++kt) {
        v16bf a1 = load_afrag(h1l, HSTRIDE_DW, mrow, g, kt);
        #pragma unroll
        for (int q = 0; q < NTW; ++q) {
          v16bf wb = load_btile(wblk, T_IH1 + (nt0 + q) * 8 + kt, lane);
          acc[q] = wmma_bf16(a1, wb, acc[q]);
        }
        v16bf a2 = load_afrag(h2l, HSTRIDE_DW, mrow, g, kt);
        #pragma unroll
        for (int q = 0; q < NTW; ++q) {
          v16bf wb = load_btile(wblk, T_HH1 + (nt0 + q) * 8 + kt, lane);
          acc[q] = wmma_bf16(a2, wb, acc[q]);
        }
      }
      #pragma unroll
      for (int q = 0; q < NTW; ++q)
        #pragma unroll
        for (int v = 0; v < 8; ++v)
          gates[(v + 8 * g) * 1024 + (nt0 + q) * 16 + mrow] = acc[q][v] + bias1[q];
    }
    __syncthreads();

    // ---- pointwise layer 1 -> h2, c2 ----
    {
      float iv[8], fv[8], gv[8], ov[8], hv[8];
      *(float4*)&iv[0] = *(const float4*)&gates[pm * 1024 + pj];
      *(float4*)&iv[4] = *(const float4*)&gates[pm * 1024 + pj + 4];
      *(float4*)&fv[0] = *(const float4*)&gates[pm * 1024 + 256 + pj];
      *(float4*)&fv[4] = *(const float4*)&gates[pm * 1024 + 256 + pj + 4];
      *(float4*)&gv[0] = *(const float4*)&gates[pm * 1024 + 512 + pj];
      *(float4*)&gv[4] = *(const float4*)&gates[pm * 1024 + 512 + pj + 4];
      *(float4*)&ov[0] = *(const float4*)&gates[pm * 1024 + 768 + pj];
      *(float4*)&ov[4] = *(const float4*)&gates[pm * 1024 + 768 + pj + 4];
      #pragma unroll
      for (int u = 0; u < 8; ++u) {
        float c = sigm(fv[u]) * c2[u] + sigm(iv[u]) * tanhf(gv[u]);
        c2[u] = c;
        hv[u] = sigm(ov[u]) * tanhf(c);
      }
      uint4 packed;
      packed.x = (unsigned)f2bf(hv[0]) | ((unsigned)f2bf(hv[1]) << 16);
      packed.y = (unsigned)f2bf(hv[2]) | ((unsigned)f2bf(hv[3]) << 16);
      packed.z = (unsigned)f2bf(hv[4]) | ((unsigned)f2bf(hv[5]) << 16);
      packed.w = (unsigned)f2bf(hv[6]) | ((unsigned)f2bf(hv[7]) << 16);
      *(uint4*)&h2l[pm * HSTRIDE + pj] = packed;
    }
    __syncthreads();

    // ---- output projection: out = h2 @ Wout^T + bout  (wave 0) ----
    if (wave == 0) {
      #pragma unroll 1
      for (int m = 0; m < 16; ++m) {
        uint4 hp = *(const uint4*)&h2l[m * HSTRIDE + lane * 8];
        float p = bf2f((unsigned short)(hp.x & 0xFFFF))        * wout_r[0]
                + bf2f((unsigned short)(hp.x >> 16))           * wout_r[1]
                + bf2f((unsigned short)(hp.y & 0xFFFF))        * wout_r[2]
                + bf2f((unsigned short)(hp.y >> 16))           * wout_r[3]
                + bf2f((unsigned short)(hp.z & 0xFFFF))        * wout_r[4]
                + bf2f((unsigned short)(hp.z >> 16))           * wout_r[5]
                + bf2f((unsigned short)(hp.w & 0xFFFF))        * wout_r[6]
                + bf2f((unsigned short)(hp.w >> 16))           * wout_r[7];
        #pragma unroll
        for (int off = 16; off > 0; off >>= 1) p += __shfl_xor(p, off, 32);
        if (lane == 0) {
          float o = p + boutv;
          out[(size_t)(b0 + m) * SS + t] = o;
          pout[m] = o;
        }
      }
    }
    __syncthreads();   // pout visible for next step's P0
  }
}

extern "C" void kernel_launch(void* const* d_in, const int* in_sizes, int n_in,
                              void* d_out, int out_size, void* d_ws, size_t ws_size,
                              hipStream_t stream) {
  (void)in_sizes; (void)n_in; (void)out_size; (void)ws_size;
  const float* x    = (const float*)d_in[0];
  const float* Wih0 = (const float*)d_in[1];
  const float* Whh0 = (const float*)d_in[2];
  const float* bih0 = (const float*)d_in[3];
  const float* bhh0 = (const float*)d_in[4];
  const float* Wih1 = (const float*)d_in[5];
  const float* Whh1 = (const float*)d_in[6];
  const float* bih1 = (const float*)d_in[7];
  const float* bhh1 = (const float*)d_in[8];
  const float* Wout = (const float*)d_in[9];
  const float* bout = (const float*)d_in[10];

  unsigned short* wbf = (unsigned short*)d_ws;
  float* fbuf = (float*)((char*)d_ws + (size_t)W_ELEMS * 2);

  int totalPrep = W_ELEMS + 2 * G4H + HH + 1;
  lstm_prep_kernel<<<(totalPrep + 255) / 256, 256, 0, stream>>>(
      Wih0, Whh0, Wih1, Whh1, bih0, bhh0, bih1, bhh1, Wout, bout, wbf, fbuf);

  size_t shmem = 16 * 1024 * 4            // gates f32
               + 2 * 16 * HSTRIDE * 2     // h1, h2 bf16 (padded rows)
               + 16 * 34 * 2              // input tile bf16 (padded)
               + 16 * 4;                  // prev_out
  lstm_scan_kernel<<<4, TPB, shmem, stream>>>(x, wbf, fbuf, (float*)d_out);
}
